// HDRFusionNet_52853867545294
// MI455X (gfx1250) — compile-verified, matrix operand order checked by reading, not run on hardware
//
#include <hip/hip_runtime.h>
#include <hip/hip_bf16.h>

// ---------------------------------------------------------------------------
// HDRFusionNet for MI455X (gfx1250, wave32, WMMA + TDM).
// v3.1 (fix): tensor_load_to_lds on this toolchain is the 6-arg form
//   (u32x4 g0, i32x8 g1, i32x4 g2, i32x4 g3, i32x8 extra, i32 cpol).
//   - conv2 / sf1 / sf2 as 9-tap shifted WMMA GEMMs with a wave-uniform
//     interior fast path (no boundary cndmask/nops for ~85% of tiles),
//   - in_proj / out_proj via generic WMMA GEMM,
//   - FFN (dominant FLOPs) fused with 16x2048 f16 hidden in LDS; the
//     16x64 A-tile is DMA'd into LDS by the Tensor Data Mover
//     (tensor_load_to_lds + s_wait_tensorcnt) and shared by all 8 waves.
// Channel-last f16 layout throughout the middle of the net (token order ==
// pixel order) removes all NCHW<->token repacks.
// ---------------------------------------------------------------------------

typedef __attribute__((ext_vector_type(16))) _Float16 v16h;
typedef __attribute__((ext_vector_type(8)))  _Float16 v8h;
typedef __attribute__((ext_vector_type(8)))  float    v8f;
typedef __attribute__((ext_vector_type(4)))  unsigned int u32x4;
typedef __attribute__((ext_vector_type(8)))  int      i32x8;
typedef __attribute__((ext_vector_type(4)))  int      i32x4;

#define HH 224
#define WW 224
#define HWTOT (HH * WW)      // 50176
#define NTOK  (2 * HWTOT)    // 100352 tokens == pixels (S=2, N=HW)
#define EDIM  64
#define DFF   2048

// ---- WMMA helpers (layouts per CDNA5 ISA 7.12.2) ---------------------------
// A 16x32 f16 frag, lane = 16*g + r: row m = r; halves 0..7 = K {8g..8g+7},
// halves 8..15 = K {16+8g..16+8g+7}  ->  two 16-byte loads at p and p+16.
// B frags from row-major [N,K] weights use the identical per-lane pattern.
__device__ __forceinline__ v16h load_frag16(const _Float16* p) {
  v8h lo = *(const v8h*)(p);
  v8h hi = *(const v8h*)(p + 16);
  return __builtin_shufflevector(lo, hi, 0, 1, 2, 3, 4, 5, 6, 7,
                                 8, 9, 10, 11, 12, 13, 14, 15);
}

__device__ __forceinline__ v8f wmma_f16(v16h a, v16h b, v8f c) {
  return __builtin_amdgcn_wmma_f32_16x16x32_f16(
      /*neg_a=*/false, a, /*neg_b=*/false, b,
      /*c_mod=*/(short)0, c, /*reuse_a=*/false, /*reuse_b=*/false);
}

__device__ __forceinline__ float wave_sum32(float v) {
#pragma unroll
  for (int off = 16; off > 0; off >>= 1) v += __shfl_xor(v, off, 32);
  return v;
}

// ---- f32 -> f16 copy -------------------------------------------------------
__global__ void cvt_f16_kernel(const float* __restrict__ src,
                               _Float16* __restrict__ dst, int n) {
  int i = blockIdx.x * blockDim.x + threadIdx.x;
  if (i < n) dst[i] = (_Float16)src[i];
}

// ---- OIHW conv weights -> per-tap f16 [9][CoPad][Ci] (zero-padded rows) ----
__global__ void repack_tap_kernel(const float* __restrict__ w,
                                  _Float16* __restrict__ wt,
                                  int Co, int CoPad, int Ci) {
  int gid = blockIdx.x * blockDim.x + threadIdx.x;
  int total = 9 * CoPad * Ci;
  if (gid >= total) return;
  int ci = gid % Ci;
  int co = (gid / Ci) % CoPad;
  int tap = gid / (Ci * CoPad);
  float v = 0.f;
  if (co < Co) v = w[((size_t)co * Ci + ci) * 9 + tap];
  wt[gid] = (_Float16)v;
}

// ---- conv1: 3->64, NCHW input, channel-last f16 output, relu ---------------
__global__ void conv1_pm_kernel(const float* __restrict__ x,
                                const float* __restrict__ wgt,
                                const float* __restrict__ bias,
                                _Float16* __restrict__ outH) {
  int gid = blockIdx.x * blockDim.x + threadIdx.x;
  if (gid >= NTOK * EDIM) return;
  int co = gid % EDIM;
  int p  = gid / EDIM;
  int b  = p / HWTOT;
  int hw = p % HWTOT;
  int y = hw / WW, xw = hw % WW;
  const float* xb = x + (size_t)b * 3 * HWTOT;
  float acc = bias[co];
#pragma unroll
  for (int ci = 0; ci < 3; ++ci) {
    const float* ip = xb + (size_t)ci * HWTOT;
    const float* wp = wgt + ((size_t)co * 3 + ci) * 9;
#pragma unroll
    for (int ky = 0; ky < 3; ++ky) {
      int yy = y + ky - 1;
      if (yy < 0 || yy >= HH) continue;
#pragma unroll
      for (int kx = 0; kx < 3; ++kx) {
        int xx = xw + kx - 1;
        if (xx < 0 || xx >= WW) continue;
        acc = fmaf(ip[yy * WW + xx], wp[ky * 3 + kx], acc);
      }
    }
  }
  acc = acc > 0.f ? acc : 0.f;
  outH[gid] = (_Float16)acc;
}

// ---- 3x3 SAME conv as 9-tap WMMA GEMM over channel-last f16 ----------------
// inp: [NTOK, K] f16; wtap: [9][NPad][K] f16 (NPad = Ntiles*16, zero-padded).
// One 16(pixel) x 16(outch) tile per wave. Interior tiles (~85%) take a
// mask-free fast path; border tiles select A frags to zero per lane
// (EXEC stays all-ones for WMMA either way).
__global__ void conv_tap_wmma_kernel(const _Float16* __restrict__ inp, int K,
                                     const _Float16* __restrict__ wtap,
                                     const float* __restrict__ bias, int ncreal,
                                     float* __restrict__ outF,
                                     _Float16* __restrict__ outH,
                                     int Ntiles, int tiles, int act) {
  int wave = threadIdx.x >> 5;
  int lane = threadIdx.x & 31;
  int tile = blockIdx.x * 8 + wave;
  if (tile >= tiles) return;          // wave-uniform
  int mt = tile / Ntiles, nt = tile % Ntiles;
  int gsel = lane >> 4, r = lane & 15;
  int NPad = Ntiles * 16;
  int n0 = nt * 16;
  int m0 = mt * 16;
  int p = m0 + r;                     // pixel handled by this lane-pair
  v8f c = {};

  // wave-uniform interior test for the whole 16-pixel tile
  int hw0 = m0 % HWTOT;
  int y0 = hw0 / WW, x0 = hw0 % WW;
  bool interior = (x0 >= 1) && (x0 <= WW - 17) && (y0 >= 1) && (y0 <= HH - 2);

  if (interior) {
#pragma unroll
    for (int tap = 0; tap < 9; ++tap) {
      int dy = tap / 3 - 1, dx = tap % 3 - 1;
      const _Float16* ar = inp + (size_t)(p + dy * WW + dx) * K + 8 * gsel;
      const _Float16* br = wtap + ((size_t)tap * NPad + n0 + r) * K + 8 * gsel;
      for (int kk = 0; kk < K; kk += 32) {
        v16h a = load_frag16(ar + kk);
        v16h bf = load_frag16(br + kk);
        c = wmma_f16(a, bf, c);
      }
    }
  } else {
    int b = p / HWTOT;
    int hw = p % HWTOT;
    int y = hw / WW, xw = hw % WW;
    v16h az = {};
#pragma unroll
    for (int tap = 0; tap < 9; ++tap) {
      int ky = tap / 3, kx = tap % 3;
      int yy = y + ky - 1, xx = xw + kx - 1;
      bool valid = (yy >= 0) && (yy < HH) && (xx >= 0) && (xx < WW);
      int srow = valid ? (b * HWTOT + yy * WW + xx) : p;  // clamp to safe addr
      const _Float16* ar = inp + (size_t)srow * K + 8 * gsel;
      const _Float16* br = wtap + ((size_t)tap * NPad + n0 + r) * K + 8 * gsel;
      for (int kk = 0; kk < K; kk += 32) {
        v16h a = load_frag16(ar + kk);
        a = valid ? a : az;
        v16h bf = load_frag16(br + kk);
        c = wmma_f16(a, bf, c);
      }
    }
  }

  int col = n0 + r;
  float bb = (col < ncreal) ? bias[col] : 0.f;
#pragma unroll
  for (int i = 0; i < 8; ++i) {
    int m = m0 + i + 8 * gsel;
    float v = c[i] + bb;
    if (act == 1) v = v > 0.f ? v : 0.f;
    if (outF) outF[(size_t)m * NPad + col] = v;
    if (outH) outH[(size_t)m * NPad + col] = (_Float16)v;
  }
}

// ---- generic WMMA GEMM: C[M,N] = A[M,K] @ W[N,K]^T + bias ------------------
__global__ void gemm_wmma_kernel(const _Float16* __restrict__ A, int lda,
                                 const _Float16* __restrict__ Wt, int ldw,
                                 const float* __restrict__ bias,
                                 float* __restrict__ C, int ldc,
                                 int Ntiles, int tiles, int K, int act) {
  int wave = threadIdx.x >> 5;
  int lane = threadIdx.x & 31;
  int tile = blockIdx.x * 8 + wave;
  if (tile >= tiles) return;          // wave-uniform
  int mt = tile / Ntiles, nt = tile % Ntiles;
  int gsel = lane >> 4, r = lane & 15;
  const _Float16* arow = A + (size_t)(mt * 16 + r) * lda + 8 * gsel;
  const _Float16* brow = Wt + (size_t)(nt * 16 + r) * ldw + 8 * gsel;
  v8f c = {};
  for (int k0 = 0; k0 < K; k0 += 32) {
    v16h a = load_frag16(arow + k0);
    v16h b = load_frag16(brow + k0);
    c = wmma_f16(a, b, c);
  }
  float bb = bias[nt * 16 + r];
  float* crow = C + (size_t)(mt * 16 + 8 * gsel) * ldc + nt * 16 + r;
#pragma unroll
  for (int i = 0; i < 8; ++i) {
    float v = c[i] + bb;
    if (act == 1) v = v > 0.f ? v : 0.f;
    crow[(size_t)i * ldc] = v;
  }
}

// ---- attention over S=2 (per (n, head): 2x2 scores, softmax, combine) ------
__global__ void attn_kernel(const float* __restrict__ qkv,
                            _Float16* __restrict__ attn_h) {
  int idx = blockIdx.x * blockDim.x + threadIdx.x;
  if (idx >= HWTOT * 4) return;
  int h = idx & 3;
  int n = idx >> 2;
  size_t b0 = (size_t)n * 192 + h * 16;
  size_t b1 = (size_t)(HWTOT + n) * 192 + h * 16;
  float q0[16], q1[16], k0[16], k1[16], v0[16], v1[16];
#pragma unroll
  for (int d = 0; d < 16; ++d) {
    q0[d] = qkv[b0 + d];       q1[d] = qkv[b1 + d];
    k0[d] = qkv[b0 + 64 + d];  k1[d] = qkv[b1 + 64 + d];
    v0[d] = qkv[b0 + 128 + d]; v1[d] = qkv[b1 + 128 + d];
  }
  float s00 = 0.f, s01 = 0.f, s10 = 0.f, s11 = 0.f;
#pragma unroll
  for (int d = 0; d < 16; ++d) {
    s00 += q0[d] * k0[d]; s01 += q0[d] * k1[d];
    s10 += q1[d] * k0[d]; s11 += q1[d] * k1[d];
  }
  s00 *= 0.25f; s01 *= 0.25f; s10 *= 0.25f; s11 *= 0.25f;  // 1/sqrt(16)
  float m0 = fmaxf(s00, s01), m1 = fmaxf(s10, s11);
  float e00 = expf(s00 - m0), e01 = expf(s01 - m0);
  float e10 = expf(s10 - m1), e11 = expf(s11 - m1);
  float r0 = 1.f / (e00 + e01), r1 = 1.f / (e10 + e11);
  float p00 = e00 * r0, p01 = e01 * r0, p10 = e10 * r1, p11 = e11 * r1;
#pragma unroll
  for (int d = 0; d < 16; ++d) {
    attn_h[(size_t)n * EDIM + h * 16 + d] =
        (_Float16)(p00 * v0[d] + p01 * v1[d]);
    attn_h[(size_t)(HWTOT + n) * EDIM + h * 16 + d] =
        (_Float16)(p10 * v0[d] + p11 * v1[d]);
  }
}

// ---- residual + LayerNorm over E=64, one token per wave --------------------
__global__ void ln_kernel(const float* __restrict__ resid,
                          const float* __restrict__ proj,
                          const float* __restrict__ gam,
                          const float* __restrict__ bet,
                          float* __restrict__ out_f,
                          _Float16* __restrict__ out_h) {
  int wave = threadIdx.x >> 5, lane = threadIdx.x & 31;
  int tok = blockIdx.x * 8 + wave;
  if (tok >= NTOK) return;
  size_t base = (size_t)tok * EDIM;
  float x0 = resid[base + lane] + proj[base + lane];
  float x1 = resid[base + lane + 32] + proj[base + lane + 32];
  float mu = wave_sum32(x0 + x1) * (1.f / 64.f);
  float d0 = x0 - mu, d1 = x1 - mu;
  float var = wave_sum32(d0 * d0 + d1 * d1) * (1.f / 64.f);
  float inv = rsqrtf(var + 1e-5f);
  float y0 = gam[lane] * d0 * inv + bet[lane];
  float y1 = gam[lane + 32] * d1 * inv + bet[lane + 32];
  out_f[base + lane] = y0;
  out_f[base + lane + 32] = y1;
  out_h[base + lane] = (_Float16)y0;
  out_h[base + lane + 32] = (_Float16)y1;
}

// ---- fused FFN: relu(t1@W1^T+b1)@W2^T+b2, +residual, LN2 -------------------
// Block = 256 threads (8 waves) owns a 16-token tile. The 16x64 f16 A-tile
// is DMA'd into LDS once via the Tensor Data Mover (wave 0 issues
// tensor_load_to_lds, waits on TENSORcnt, then workgroup barrier); all
// waves read their A frags from LDS. 16x2048 f16 hidden lives in LDS.
// LDS map: [0,64K) hidden (reused as f32 partials), [64K, 64K+2K) A staging.
__global__ void ffn_fused_kernel(const _Float16* __restrict__ t1h,
                                 const float* __restrict__ t1f,
                                 const _Float16* __restrict__ w1h,
                                 const float* __restrict__ bias1,
                                 const _Float16* __restrict__ w2h,
                                 const float* __restrict__ bias2,
                                 const float* __restrict__ g2,
                                 const float* __restrict__ b2v,
                                 float* __restrict__ t2f,
                                 _Float16* __restrict__ t2h) {
  extern __shared__ char smem[];
  _Float16* hid = (_Float16*)smem;               // [16][2048] f16 = 64KB
  _Float16* ast = (_Float16*)(smem + 65536);     // [16][64] f16 A staging
  int wave = threadIdx.x >> 5, lane = threadIdx.x & 31;
  int gsel = lane >> 4, r = lane & 15;
  int m0 = blockIdx.x * 16;

  // --- TDM: DMA the 16x64 f16 A tile (t1h rows m0..m0+15) into LDS ---
  // D# group0: count=1, lds_addr=65536 (dynamic LDS assumed at offset 0),
  //            global_addr = &t1h[m0*64], type=2 ("image").
  // D# group1: data_size=1 (2B), tensor_dim0=64, tensor_dim1=16,
  //            tile_dim0=64, tile_dim1=16, tensor_dim0_stride=64.
  if (wave == 0) {
    unsigned long long ga = (unsigned long long)(uintptr_t)(t1h + (size_t)m0 * EDIM);
    u32x4 g0 = {1u, 65536u, (unsigned)(ga & 0xFFFFFFFFu),
                (unsigned)((ga >> 32) & 0x01FFFFFFu) | 0x80000000u};
    i32x8 g1 = {(int)0x00010000,        // data_size=1 (2 bytes)
                (int)(64u << 16),       // tensor_dim0[15:0] in [31:16]
                (int)(16u << 16),       // tensor_dim0 hi=0 | tensor_dim1 lo
                (int)(64u << 16),       // tensor_dim1 hi=0 | tile_dim0=64
                (int)16,                // tile_dim1=16 (tile_dim2=0)
                (int)64,                // tensor_dim0_stride lo32
                0, 0};
    i32x4 gz4 = {0, 0, 0, 0};
    i32x8 gz8 = {0, 0, 0, 0, 0, 0, 0, 0};
    __builtin_amdgcn_tensor_load_to_lds(g0, g1, gz4, gz4, gz8, 0);
    __builtin_amdgcn_s_wait_tensorcnt(0);
  }
  __syncthreads();

  const _Float16* arow = ast + (size_t)r * EDIM + 8 * gsel;
  v16h a0 = load_frag16(arow);
  v16h a1 = load_frag16(arow + 32);

  // --- FFN1: wave w covers hidden cols [w*256, w*256+256) ---
  for (int nt = 0; nt < 16; ++nt) {
    int n0 = wave * 256 + nt * 16;
    const _Float16* brow = w1h + (size_t)(n0 + r) * EDIM + 8 * gsel;
    v16h b0 = load_frag16(brow);
    v16h b1 = load_frag16(brow + 32);
    v8f c = {};
    c = wmma_f16(a0, b0, c);
    c = wmma_f16(a1, b1, c);
    float bb = bias1[n0 + r];
#pragma unroll
    for (int i = 0; i < 8; ++i) {
      float v = c[i] + bb;
      v = v > 0.f ? v : 0.f;
      hid[(i + 8 * gsel) * DFF + n0 + r] = (_Float16)v;
    }
  }
  __syncthreads();

  // --- FFN2: wave w covers K chunk [w*256, w*256+256), all 4 N-tiles ---
  v8f zero = {};
  v8f c2[4] = {zero, zero, zero, zero};
  int kbase = wave * 256;
  for (int ks = 0; ks < 8; ++ks) {
    int kk = kbase + ks * 32;
    v16h a = load_frag16(hid + r * DFF + kk + 8 * gsel);
#pragma unroll
    for (int nt = 0; nt < 4; ++nt) {
      const _Float16* brow = w2h + (size_t)(nt * 16 + r) * DFF + kk + 8 * gsel;
      v16h b = load_frag16(brow);
      c2[nt] = wmma_f16(a, b, c2[nt]);
    }
  }
  __syncthreads();  // all hid reads done; reuse LDS for f32 partials

  float* part = (float*)smem;        // [8][16*64] f32 = 32KB
  float* myp = part + wave * 1024;
#pragma unroll
  for (int nt = 0; nt < 4; ++nt)
#pragma unroll
    for (int i = 0; i < 8; ++i)
      myp[(i + 8 * gsel) * 64 + nt * 16 + r] = c2[nt][i];
  __syncthreads();

  for (int e = threadIdx.x; e < 1024; e += 256) {
    float s = 0.f;
#pragma unroll
    for (int w = 0; w < 8; ++w) s += part[w * 1024 + e];
    part[e] = s;
  }
  __syncthreads();

  // --- residual + LN2, channel-last output: 2 tokens per wave ---
  for (int tl = wave; tl < 16; tl += 8) {
    int tok = m0 + tl;
    float x0 = t1f[(size_t)tok * EDIM + lane] + part[tl * 64 + lane] + bias2[lane];
    float x1 = t1f[(size_t)tok * EDIM + lane + 32] + part[tl * 64 + lane + 32] +
               bias2[lane + 32];
    float mu = wave_sum32(x0 + x1) * (1.f / 64.f);
    float d0 = x0 - mu, d1 = x1 - mu;
    float var = wave_sum32(d0 * d0 + d1 * d1) * (1.f / 64.f);
    float inv = rsqrtf(var + 1e-5f);
    float y0 = g2[lane] * d0 * inv + b2v[lane];
    float y1 = g2[lane + 32] * d1 * inv + b2v[lane + 32];
    size_t base = (size_t)tok * EDIM;
    t2f[base + lane] = y0;
    t2f[base + lane + 32] = y1;
    t2h[base + lane] = (_Float16)y0;
    t2h[base + lane + 32] = (_Float16)y1;
  }
}

// ---- KPN fuse: per-pixel 3x3 filter (channel-last f2, [NTOK,16]) -----------
__global__ void kpn_kernel(const float* __restrict__ x,
                           const float* __restrict__ f2p,
                           float* __restrict__ fused) {
  int gid = blockIdx.x * blockDim.x + threadIdx.x;
  if (gid >= NTOK) return;
  int b = gid / HWTOT;
  int hw = gid % HWTOT;
  int y = hw / WW, xw = hw % WW;
  const float* xb = x + (size_t)b * 3 * HWTOT;
  float acc = 0.f;
#pragma unroll
  for (int ky = 0; ky < 3; ++ky) {
#pragma unroll
    for (int kx = 0; kx < 3; ++kx) {
      int yy = y + ky - 1, xx = xw + kx - 1;
      float sv = 0.f;
      if (yy >= 0 && yy < HH && xx >= 0 && xx < WW) {
        int p = yy * WW + xx;
        sv = xb[p] + xb[HWTOT + p] + xb[2 * HWTOT + p];
      }
      acc += f2p[(size_t)gid * 16 + (ky * 3 + kx)] * sv;
    }
  }
  fused[gid] = acc;
}

// ---- direct 3x3 SAME conv (decoder tail only), NCHW ------------------------
__global__ void conv3x3_kernel(const float* __restrict__ in,
                               const float* __restrict__ wgt,
                               const float* __restrict__ bias,
                               float* __restrict__ out,
                               int Cin, int Cout, int act) {
  int gid = blockIdx.x * blockDim.x + threadIdx.x;
  int total = 2 * Cout * HWTOT;
  if (gid >= total) return;
  int xw = gid % WW;
  int y  = (gid / WW) % HH;
  int co = (gid / HWTOT) % Cout;
  int b  = gid / (HWTOT * Cout);
  const float* inb = in + (size_t)b * Cin * HWTOT;
  float acc = bias[co];
  for (int ci = 0; ci < Cin; ++ci) {
    const float* ip = inb + (size_t)ci * HWTOT;
    const float* wp = wgt + ((size_t)co * Cin + ci) * 9;
#pragma unroll
    for (int ky = 0; ky < 3; ++ky) {
      int yy = y + ky - 1;
      if (yy < 0 || yy >= HH) continue;
#pragma unroll
      for (int kx = 0; kx < 3; ++kx) {
        int xx = xw + kx - 1;
        if (xx < 0 || xx >= WW) continue;
        acc = fmaf(ip[yy * WW + xx], wp[ky * 3 + kx], acc);
      }
    }
  }
  if (act == 1) acc = acc > 0.f ? acc : 0.f;
  else if (act == 2) acc = 1.f / (1.f + expf(-acc));
  out[gid] = acc;
}

// ---------------------------------------------------------------------------
extern "C" void kernel_launch(void* const* d_in, const int* in_sizes, int n_in,
                              void* d_out, int out_size, void* d_ws,
                              size_t ws_size, hipStream_t stream) {
  (void)in_sizes; (void)n_in; (void)out_size; (void)ws_size;
  const float* x       = (const float*)d_in[0];
  const float* enc_w1  = (const float*)d_in[1];
  const float* enc_b1  = (const float*)d_in[2];
  const float* enc_w2  = (const float*)d_in[3];
  const float* enc_b2  = (const float*)d_in[4];
  const float* ipw     = (const float*)d_in[5];
  const float* ipb     = (const float*)d_in[6];
  const float* opw     = (const float*)d_in[7];
  const float* opb     = (const float*)d_in[8];
  const float* ln1g    = (const float*)d_in[9];
  const float* ln1b    = (const float*)d_in[10];
  const float* fw1     = (const float*)d_in[11];
  const float* fb1     = (const float*)d_in[12];
  const float* fw2     = (const float*)d_in[13];
  const float* fb2     = (const float*)d_in[14];
  const float* ln2g    = (const float*)d_in[15];
  const float* ln2b    = (const float*)d_in[16];
  const float* sf_w1   = (const float*)d_in[17];
  const float* sf_b1   = (const float*)d_in[18];
  const float* sf_w2   = (const float*)d_in[19];
  const float* sf_b2   = (const float*)d_in[20];
  const float* dec_w1  = (const float*)d_in[21];
  const float* dec_b1  = (const float*)d_in[22];
  const float* dec_w2  = (const float*)d_in[23];
  const float* dec_b2  = (const float*)d_in[24];
  float* out = (float*)d_out;

  // workspace arena (256B aligned slices)
  char* ws = (char*)d_ws;
  size_t off = 0;
  auto alloc = [&](size_t bytes) {
    size_t o = off;
    off += (bytes + 255) & ~(size_t)255;
    return o;
  };
  _Float16* e1h    = (_Float16*)(ws + alloc((size_t)NTOK * EDIM * 2));
  float*    t_f    = (float*)(ws + alloc((size_t)NTOK * EDIM * 4));
  _Float16* t_h    = (_Float16*)(ws + alloc((size_t)NTOK * EDIM * 2));
  float*    qkv    = (float*)(ws + alloc((size_t)NTOK * 192 * 4));
  _Float16* attn_h = (_Float16*)(ws + alloc((size_t)NTOK * EDIM * 2));
  float*    proj   = (float*)(ws + alloc((size_t)NTOK * EDIM * 4));
  float*    t1_f   = (float*)(ws + alloc((size_t)NTOK * EDIM * 4));
  _Float16* t1_h   = (_Float16*)(ws + alloc((size_t)NTOK * EDIM * 2));
  float*    t2_f   = (float*)(ws + alloc((size_t)NTOK * EDIM * 4));
  _Float16* t2_h   = (_Float16*)(ws + alloc((size_t)NTOK * EDIM * 2));
  _Float16* f1_h   = (_Float16*)(ws + alloc((size_t)NTOK * 32 * 2));
  float*    f2_f   = (float*)(ws + alloc((size_t)NTOK * 16 * 4));
  float*    fusedb = (float*)(ws + alloc((size_t)NTOK * 4));
  float*    dbuf   = (float*)(ws + alloc((size_t)NTOK * EDIM * 4));
  _Float16* ipw_h  = (_Float16*)(ws + alloc((size_t)192 * 64 * 2));
  _Float16* opw_h  = (_Float16*)(ws + alloc((size_t)64 * 64 * 2));
  _Float16* w1_h   = (_Float16*)(ws + alloc((size_t)DFF * 64 * 2));
  _Float16* w2_h   = (_Float16*)(ws + alloc((size_t)64 * DFF * 2));
  _Float16* w2tap  = (_Float16*)(ws + alloc((size_t)9 * 64 * 64 * 2));
  _Float16* sf1tap = (_Float16*)(ws + alloc((size_t)9 * 32 * 64 * 2));
  _Float16* sf2tap = (_Float16*)(ws + alloc((size_t)9 * 16 * 32 * 2));

  const int TB = 256;
  auto grids = [](long long n, int tb) { return (unsigned)((n + tb - 1) / tb); };

  // weight conversion / per-tap repack (tiny)
  cvt_f16_kernel<<<grids(192 * 64, TB), TB, 0, stream>>>(ipw, ipw_h, 192 * 64);
  cvt_f16_kernel<<<grids(64 * 64, TB), TB, 0, stream>>>(opw, opw_h, 64 * 64);
  cvt_f16_kernel<<<grids(DFF * 64, TB), TB, 0, stream>>>(fw1, w1_h, DFF * 64);
  cvt_f16_kernel<<<grids(64 * DFF, TB), TB, 0, stream>>>(fw2, w2_h, 64 * DFF);
  repack_tap_kernel<<<grids(9 * 64 * 64, TB), TB, 0, stream>>>(enc_w2, w2tap, 64, 64, 64);
  repack_tap_kernel<<<grids(9 * 32 * 64, TB), TB, 0, stream>>>(sf_w1, sf1tap, 32, 32, 64);
  repack_tap_kernel<<<grids(9 * 16 * 32, TB), TB, 0, stream>>>(sf_w2, sf2tap, 9, 16, 32);

  // conv1 (3->64) -> channel-last f16
  conv1_pm_kernel<<<grids((long long)NTOK * EDIM, TB), TB, 0, stream>>>(
      x, enc_w1, enc_b1, e1h);

  // conv2 (64->64) as 9-tap WMMA GEMM; epilogue = transformer input t (f32+f16)
  {
    int tiles = (NTOK / 16) * 4;
    conv_tap_wmma_kernel<<<grids(tiles, 8), TB, 0, stream>>>(
        e1h, 64, w2tap, enc_b2, 64, t_f, t_h, 4, tiles, 1);
  }

  // qkv = t @ in_proj_w^T + b   (M=100352, N=192, K=64)
  {
    int tiles = (NTOK / 16) * 12;
    gemm_wmma_kernel<<<grids(tiles, 8), TB, 0, stream>>>(
        t_h, EDIM, ipw_h, EDIM, ipb, qkv, 192, 12, tiles, 64, 0);
  }

  // attention over S=2
  attn_kernel<<<grids((long long)HWTOT * 4, TB), TB, 0, stream>>>(qkv, attn_h);

  // out_proj (M=100352, N=64, K=64)
  {
    int tiles = (NTOK / 16) * 4;
    gemm_wmma_kernel<<<grids(tiles, 8), TB, 0, stream>>>(
        attn_h, EDIM, opw_h, EDIM, opb, proj, EDIM, 4, tiles, 64, 0);
  }

  // t1 = LN1(t + attn_out)
  ln_kernel<<<grids(NTOK, 8), TB, 0, stream>>>(t_f, proj, ln1g, ln1b, t1_f, t1_h);

  // fused FFN + residual + LN2 -> enc2 channel-last (f32 + f16)
  // LDS: 64KB hidden + 2KB TDM-staged A tile
  ffn_fused_kernel<<<NTOK / 16, TB, 16 * DFF * 2 + 2048, stream>>>(
      t1_h, t1_f, w1_h, fb1, w2_h, fb2, ln2g, ln2b, t2_f, t2_h);

  // sf1 (64->32) tap-GEMM -> f16 channel-last
  {
    int tiles = (NTOK / 16) * 2;
    conv_tap_wmma_kernel<<<grids(tiles, 8), TB, 0, stream>>>(
        t2_h, 64, sf1tap, sf_b1, 32, (float*)nullptr, f1_h, 2, tiles, 0);
  }
  // sf2 (32->9, padded to 16) tap-GEMM -> f32 channel-last
  {
    int tiles = (NTOK / 16) * 1;
    conv_tap_wmma_kernel<<<grids(tiles, 8), TB, 0, stream>>>(
        f1_h, 32, sf2tap, sf_b2, 9, f2_f, (_Float16*)nullptr, 1, tiles, 0);
  }

  // KPN fuse
  kpn_kernel<<<grids(NTOK, TB), TB, 0, stream>>>(x, f2_f, fusedb);

  // decoder (tiny, VALU)
  conv3x3_kernel<<<grids((long long)NTOK * EDIM, TB), TB, 0, stream>>>(
      fusedb, dec_w1, dec_b1, dbuf, 1, 64, 1);
  conv3x3_kernel<<<grids((long long)NTOK, TB), TB, 0, stream>>>(
      dbuf, dec_w2, dec_b2, out, 64, 1, 2);
}
